// TLDetector_52639119180042
// MI455X (gfx1250) — matrix operand order repaired and checked by reading, not compiled, not run on hardware
//
#include <hip/hip_runtime.h>
#include <math.h>

// ---------------- constants from the reference ----------------
#define STRIDE_    16
#define HF_        160
#define WF_        160
#define A_NUM      15
#define NANCH      (A_NUM * HF_ * WF_)       // 384000
#define PRE_NMS_   3000
#define RPN_KEEP_  300
#define OBJ_TH_    0.2f
#define MIN_SIZE_  8.0f
#define RPN_NMS_TH_ 0.7f
#define G_         7
#define OUT_DIM_   10
#define FEAT_DIM_  490                        // OUT_DIM*G*G
#define HID_       2048
#define NUM_FG_    3
#define SCORE_TH_  0.3f
#define RCNN_NMS_TH_ 0.5f
#define FINAL_KEEP_ 100
#define CLIP_HI_   2559.0f                    // IM-1

// histogram select
#define NBINS_     8192
#define BIN0_      0x3E4CC                    // bits(0.2f) >> 12

// GEMM tiling
#define MT_        20                         // ceil(300/16)=19, padded to 20
#define KT_        16                         // 512/32
#define NT_        128                        // 2048/16

typedef __attribute__((ext_vector_type(16))) _Float16 v16h;
typedef __attribute__((ext_vector_type(8)))  float    v8f;

// ==================== RPN anchor decode + filter ====================
__global__ void rpn_decode_kernel(const float* __restrict__ cls,
                                  const float* __restrict__ bb,
                                  float* __restrict__ boxes,
                                  float* __restrict__ scores) {
  int i = blockIdx.x * blockDim.x + threadIdx.x;
  if (i >= NANCH) return;
  int a = i / (HF_ * WF_);
  int rem = i % (HF_ * WF_);
  int y = rem / WF_, x = rem % WF_;
  const float ratios[3] = {0.5f, 1.0f, 2.0f};
  const float scales[5] = {2.0f, 4.0f, 8.0f, 16.0f, 32.0f};
  float r = ratios[a / 5], s = scales[a % 5];
  float size = 16.0f * s;
  float wA = size * sqrtf(1.0f / r);
  float hA = size * sqrtf(r);
  float gx = (float)x * 16.0f, gy = (float)y * 16.0f;
  float ax1 = gx - 0.5f * wA, ay1 = gy - 0.5f * hA;
  float ax2 = gx + 0.5f * wA, ay2 = gy + 0.5f * hA;
  float wa = ax2 - ax1 + 1.0f, ha = ay2 - ay1 + 1.0f;
  float cxa = ax1 + 0.5f * wa, cya = ay1 + 0.5f * ha;
  int base = ((a * 4) * HF_ + y) * WF_ + x;
  float d0 = bb[base] * 0.1f;
  float d1 = bb[base + HF_ * WF_] * 0.1f;
  float d2 = bb[base + 2 * HF_ * WF_] * 0.2f;
  float d3 = bb[base + 3 * HF_ * WF_] * 0.2f;
  float cx = d0 * wa + cxa, cy = d1 * ha + cya;
  float w = wa * expf(fminf(d2, 4.0f));
  float h = ha * expf(fminf(d3, 4.0f));
  float x1 = fminf(fmaxf(cx - 0.5f * w, 0.0f), CLIP_HI_);
  float y1 = fminf(fmaxf(cy - 0.5f * h, 0.0f), CLIP_HI_);
  float x2 = fminf(fmaxf(cx + 0.5f * w, 0.0f), CLIP_HI_);
  float y2 = fminf(fmaxf(cy + 0.5f * h, 0.0f), CLIP_HI_);
  boxes[i * 4 + 0] = x1; boxes[i * 4 + 1] = y1;
  boxes[i * 4 + 2] = x2; boxes[i * 4 + 3] = y2;
  float sc = cls[(A_NUM + a) * HF_ * WF_ + y * WF_ + x];
  float bw = x2 - x1 + 1.0f, bh = y2 - y1 + 1.0f;
  scores[i] = (bw >= MIN_SIZE_ && bh >= MIN_SIZE_ && sc > OBJ_TH_) ? sc : 0.0f;
}

// ==================== top-3000 via float-bit histogram ====================
__device__ __forceinline__ int score_bin(float s) {
  unsigned bits = __float_as_uint(s);
  int bin = (int)(bits >> 12) - BIN0_;
  return bin < 0 ? 0 : (bin > NBINS_ - 1 ? NBINS_ - 1 : bin);
}

__global__ void hist_kernel(const float* __restrict__ scores, unsigned* __restrict__ hist) {
  int i = blockIdx.x * blockDim.x + threadIdx.x;
  if (i >= NANCH) return;
  float s = scores[i];
  if (s <= 0.0f) return;
  atomicAdd(&hist[score_bin(s)], 1u);
}

__global__ void thresh_kernel(const unsigned* __restrict__ hist, int* __restrict__ meta) {
  if (threadIdx.x != 0 || blockIdx.x != 0) return;
  unsigned acc = 0;
  int T = -1; unsigned above = 0;
  for (int b = NBINS_ - 1; b >= 0; --b) {
    unsigned c = hist[b];
    if (acc + c >= PRE_NMS_) { T = b; above = acc; break; }
    acc += c;
  }
  if (T < 0) { T = -1; above = 0; }
  meta[0] = T; meta[1] = (int)above;
}

__global__ void compact_kernel(const float* __restrict__ scores, const float* __restrict__ boxes,
                               const int* __restrict__ meta, unsigned* __restrict__ cnt,
                               float* __restrict__ sel_s, float* __restrict__ sel_b) {
  int i = blockIdx.x * blockDim.x + threadIdx.x;
  if (i >= NANCH) return;
  float s = scores[i];
  if (s <= 0.0f) return;
  int bin = score_bin(s);
  int T = meta[0], above = meta[1];
  int pos = -1;
  if (bin > T) {
    pos = (int)atomicAdd(&cnt[0], 1u);
  } else if (bin == T) {
    int p = above + (int)atomicAdd(&cnt[1], 1u);
    if (p < PRE_NMS_) pos = p;
  }
  if (pos >= 0 && pos < PRE_NMS_) {
    sel_s[pos] = s;
    sel_b[pos * 4 + 0] = boxes[i * 4 + 0];
    sel_b[pos * 4 + 1] = boxes[i * 4 + 1];
    sel_b[pos * 4 + 2] = boxes[i * 4 + 2];
    sel_b[pos * 4 + 3] = boxes[i * 4 + 3];
  }
}

// ==================== greedy NMS (single workgroup) ====================
__global__ void rpn_nms_kernel(const float* __restrict__ sel_b, const float* __restrict__ sel_s,
                               float* __restrict__ rois, float* __restrict__ validf) {
  __shared__ float ls[PRE_NMS_];
  __shared__ float rv[256];
  __shared__ int   ri[256];
  __shared__ float bbx[4];
  __shared__ int   curi;
  int tid = threadIdx.x;
  for (int j = tid; j < PRE_NMS_; j += 256) ls[j] = sel_s[j];
  __syncthreads();
  for (int k = 0; k < RPN_KEEP_; ++k) {
    float best = -1e30f; int bi = -1;
    for (int j = tid; j < PRE_NMS_; j += 256) {
      float v = ls[j];
      if (v > best) { best = v; bi = j; }
    }
    rv[tid] = best; ri[tid] = bi;
    __syncthreads();
    for (int off = 128; off > 0; off >>= 1) {
      if (tid < off) {
        float vo = rv[tid + off]; int io = ri[tid + off];
        if (vo > rv[tid] || (vo == rv[tid] && io >= 0 && (ri[tid] < 0 || io < ri[tid]))) {
          rv[tid] = vo; ri[tid] = io;
        }
      }
      __syncthreads();
    }
    if (tid == 0) {
      int i = ri[0];
      bool ok = (i >= 0) && (rv[0] > 0.0f);
      curi = ok ? i : -1;
      if (ok) {
        bbx[0] = sel_b[i * 4 + 0]; bbx[1] = sel_b[i * 4 + 1];
        bbx[2] = sel_b[i * 4 + 2]; bbx[3] = sel_b[i * 4 + 3];
        rois[k * 4 + 0] = bbx[0]; rois[k * 4 + 1] = bbx[1];
        rois[k * 4 + 2] = bbx[2]; rois[k * 4 + 3] = bbx[3];
        validf[k] = 1.0f;
        ls[i] = -1.0f;
      } else {
        rois[k * 4 + 0] = 0.0f; rois[k * 4 + 1] = 0.0f;
        rois[k * 4 + 2] = 0.0f; rois[k * 4 + 3] = 0.0f;
        validf[k] = 0.0f;
      }
    }
    __syncthreads();
    int ci = curi;
    if (ci >= 0) {
      float bx1 = bbx[0], by1 = bbx[1], bx2 = bbx[2], by2 = bbx[3];
      float areai = fmaxf(bx2 - bx1 + 1.0f, 0.0f) * fmaxf(by2 - by1 + 1.0f, 0.0f);
      for (int j = tid; j < PRE_NMS_; j += 256) {
        if (ls[j] <= 0.0f) continue;
        float jx1 = sel_b[j * 4 + 0], jy1 = sel_b[j * 4 + 1];
        float jx2 = sel_b[j * 4 + 2], jy2 = sel_b[j * 4 + 3];
        float areaj = fmaxf(jx2 - jx1 + 1.0f, 0.0f) * fmaxf(jy2 - jy1 + 1.0f, 0.0f);
        float ix1 = fmaxf(bx1, jx1), iy1 = fmaxf(by1, jy1);
        float ix2 = fminf(bx2, jx2), iy2 = fminf(by2, jy2);
        float inter = fmaxf(ix2 - ix1 + 1.0f, 0.0f) * fmaxf(iy2 - iy1 + 1.0f, 0.0f);
        float iou = inter / fmaxf(areai + areaj - inter, 1e-6f);
        if (iou > RPN_NMS_TH_) ls[j] = -1.0f;
      }
    }
    __syncthreads();
  }
}

// ==================== PSROI align ====================
__global__ void psroi_kernel(const float* __restrict__ ft, const float* __restrict__ rois,
                             const float* __restrict__ validf, float* __restrict__ pooled) {
  int idx = blockIdx.x * blockDim.x + threadIdx.x;
  if (idx >= RPN_KEEP_ * FEAT_DIM_) return;
  int r = idx / FEAT_DIM_, c = idx % FEAT_DIM_;
  int gy = (c / G_) % G_, gx = c % G_;
  float x1 = rois[r * 4 + 0] / 16.0f, y1 = rois[r * 4 + 1] / 16.0f;
  float x2 = rois[r * 4 + 2] / 16.0f, y2 = rois[r * 4 + 3] / 16.0f;
  float bw = fmaxf(x2 - x1, 0.1f) / (float)G_;
  float bh = fmaxf(y2 - y1, 0.1f) / (float)G_;
  const float* f = ft + c * HF_ * WF_;
  float acc = 0.0f;
  for (int sy = 0; sy < 2; ++sy) {
    for (int sx = 0; sx < 2; ++sx) {
      float fy = (float)gy + ((float)sy + 0.5f) * 0.5f;
      float fx = (float)gx + ((float)sx + 0.5f) * 0.5f;
      float ys = fminf(fmaxf(y1 + fy * bh, 0.0f), (float)(HF_ - 1));
      float xs = fminf(fmaxf(x1 + fx * bw, 0.0f), (float)(WF_ - 1));
      float y0 = floorf(ys), x0 = floorf(xs);
      float wy = ys - y0, wx = xs - x0;
      int y0i = (int)y0, x0i = (int)x0;
      int y1i = min(y0i + 1, HF_ - 1), x1i = min(x0i + 1, WF_ - 1);
      float v00 = f[y0i * WF_ + x0i], v01 = f[y0i * WF_ + x1i];
      float v10 = f[y1i * WF_ + x0i], v11 = f[y1i * WF_ + x1i];
      acc += v00 * (1.0f - wy) * (1.0f - wx) + v01 * (1.0f - wy) * wx +
             v10 * wy * (1.0f - wx) + v11 * wy * wx;
    }
  }
  pooled[idx] = 0.25f * acc * validf[r];
}

// ==================== pack A (feats) into WMMA 16-bit A layout ====================
// Lane L: half=L/16, m=L%16; VGPR v holds K = kbase, kbase+1,
// kbase = 2v + (v>=4 ? 8 : 0) + 8*half  (per ISA 16-bit A 16x32 table)
__global__ void packA_kernel(const float* __restrict__ pooled, unsigned* __restrict__ Apack) {
  int idx = blockIdx.x * blockDim.x + threadIdx.x;
  if (idx >= MT_ * KT_ * 32 * 8) return;
  int v    = idx & 7;
  int lane = (idx >> 3) & 31;
  int kt   = (idx >> 8) & 15;
  int mt   = idx >> 12;
  int half = lane >> 4;
  int m  = mt * 16 + (lane & 15);
  int kb = kt * 32 + 2 * v + (v >= 4 ? 8 : 0) + 8 * half;
  float f0 = (m < RPN_KEEP_ && kb     < FEAT_DIM_) ? pooled[m * FEAT_DIM_ + kb]     : 0.0f;
  float f1 = (m < RPN_KEEP_ && kb + 1 < FEAT_DIM_) ? pooled[m * FEAT_DIM_ + kb + 1] : 0.0f;
  union { _Float16 h[2]; unsigned u; } pk;
  pk.h[0] = (_Float16)f0; pk.h[1] = (_Float16)f1;
  Apack[idx] = pk.u;
}

// ==================== pack B (w1) into WMMA 16-bit B layout ====================
// Lane L: khalf=L/16, n=L%16; VGPR v holds K = 2v + (v>=4 ? 8 : 0) + 8*khalf, +1 for column n
__global__ void packB_kernel(const float* __restrict__ w1, unsigned* __restrict__ Bpack) {
  int idx = blockIdx.x * blockDim.x + threadIdx.x;
  if (idx >= NT_ * KT_ * 32 * 8) return;
  int v    = idx & 7;
  int lane = (idx >> 3) & 31;
  int kt   = (idx >> 8) & 15;
  int nt   = idx >> 12;
  int k = kt * 32 + 2 * v + (v >= 4 ? 8 : 0) + 8 * (lane >> 4);
  int n = nt * 16 + (lane & 15);
  float f0 = (k     < FEAT_DIM_) ? w1[k * HID_ + n]       : 0.0f;
  float f1 = (k + 1 < FEAT_DIM_) ? w1[(k + 1) * HID_ + n] : 0.0f;
  union { _Float16 h[2]; unsigned u; } pk;
  pk.h[0] = (_Float16)f0; pk.h[1] = (_Float16)f1;
  Bpack[idx] = pk.u;
}

// ==================== FC1 GEMM with v_wmma_f32_16x16x32_f16 ====================
// 2560 tiles (20 M-tiles x 128 N-tiles), 8 waves/block, 1 tile/wave, K = 16 steps
__global__ void gemm_fc1_kernel(const unsigned* __restrict__ Apack,
                                const unsigned* __restrict__ Bpack,
                                const float* __restrict__ b1,
                                float* __restrict__ h) {
  int wave = threadIdx.x >> 5;
  int lane = threadIdx.x & 31;
  int tile = blockIdx.x * 8 + wave;     // grid sized exactly: tile < 2560
  int mt = tile >> 7;
  int nt = tile & 127;
  v8f acc = {};
#pragma unroll
  for (int kt = 0; kt < KT_; ++kt) {
    v16h a = *(const v16h*)(Apack + (((mt * KT_ + kt) * 32 + lane) << 3));
    v16h b = *(const v16h*)(Bpack + (((nt * KT_ + kt) * 32 + lane) << 3));
    acc = __builtin_amdgcn_wmma_f32_16x16x32_f16(false, a, false, b, (short)0, acc,
                                                 false, false);
  }
  int half = lane >> 4;
  int n = nt * 16 + (lane & 15);
  float bias = b1[n];
#pragma unroll
  for (int r = 0; r < 8; ++r) {
    int m = mt * 16 + r + 8 * half;     // C layout: VGPR r -> M = r + 8*half, N = lane%16
    if (m < RPN_KEEP_) h[m * HID_ + n] = fmaxf(acc[r] + bias, 0.0f);
  }
}

// ==================== FC2 heads + softmax ====================
__global__ void fc2_kernel(const float* __restrict__ h,
                           const float* __restrict__ w_cls, const float* __restrict__ b_cls,
                           const float* __restrict__ w_bbox, const float* __restrict__ b_bbox,
                           float* __restrict__ cls_sm, float* __restrict__ bboxo) {
  int r = blockIdx.x;
  int tid = threadIdx.x;
  float acc[20];
#pragma unroll
  for (int j = 0; j < 20; ++j) acc[j] = 0.0f;
  for (int k = tid; k < HID_; k += 256) {
    float hv = h[r * HID_ + k];
#pragma unroll
    for (int j = 0; j < 4; ++j)  acc[j]     += hv * w_cls[k * 4 + j];
#pragma unroll
    for (int j = 0; j < 16; ++j) acc[4 + j] += hv * w_bbox[k * 16 + j];
  }
  __shared__ float red[256];
  __shared__ float tot[20];
  for (int j = 0; j < 20; ++j) {
    red[tid] = acc[j];
    __syncthreads();
    for (int off = 128; off > 0; off >>= 1) {
      if (tid < off) red[tid] += red[tid + off];
      __syncthreads();
    }
    if (tid == 0) tot[j] = red[0];
    __syncthreads();
  }
  if (tid == 0) {
    float l[4], m = -1e30f;
    for (int j = 0; j < 4; ++j) { l[j] = tot[j] + b_cls[j]; m = fmaxf(m, l[j]); }
    float e[4], sum = 0.0f;
    for (int j = 0; j < 4; ++j) { e[j] = expf(l[j] - m); sum += e[j]; }
    for (int j = 0; j < 4; ++j) cls_sm[r * 4 + j] = e[j] / sum;
    for (int j = 0; j < 16; ++j) bboxo[r * 16 + j] = tot[4 + j] + b_bbox[j];
  }
}

// ==================== RCNN decode ====================
__global__ void rcnn_decode_kernel(const float* __restrict__ rois, const float* __restrict__ validf,
                                   const float* __restrict__ cls_sm, const float* __restrict__ bboxp,
                                   float* __restrict__ fb, float* __restrict__ fs,
                                   float* __restrict__ flab) {
  int i = blockIdx.x * blockDim.x + threadIdx.x;
  if (i >= RPN_KEEP_ * NUM_FG_) return;
  int r = i / NUM_FG_;
  int j = i % NUM_FG_ + 1;
  float x1 = rois[r * 4 + 0], y1 = rois[r * 4 + 1];
  float x2 = rois[r * 4 + 2], y2 = rois[r * 4 + 3];
  float wa = x2 - x1 + 1.0f, ha = y2 - y1 + 1.0f;
  float cxa = x1 + 0.5f * wa, cya = y1 + 0.5f * ha;
  float d0 = bboxp[r * 16 + j * 4 + 0] * 0.1f;
  float d1 = bboxp[r * 16 + j * 4 + 1] * 0.1f;
  float d2 = bboxp[r * 16 + j * 4 + 2] * 0.2f;
  float d3 = bboxp[r * 16 + j * 4 + 3] * 0.2f;
  float cx = d0 * wa + cxa, cy = d1 * ha + cya;
  float w = wa * expf(fminf(d2, 4.0f));
  float h = ha * expf(fminf(d3, 4.0f));
  fb[i * 4 + 0] = fminf(fmaxf(cx - 0.5f * w, 0.0f), CLIP_HI_);
  fb[i * 4 + 1] = fminf(fmaxf(cy - 0.5f * h, 0.0f), CLIP_HI_);
  fb[i * 4 + 2] = fminf(fmaxf(cx + 0.5f * w, 0.0f), CLIP_HI_);
  fb[i * 4 + 3] = fminf(fmaxf(cy + 0.5f * h, 0.0f), CLIP_HI_);
  float s = cls_sm[r * 4 + j] * validf[r];
  fs[i] = (s > SCORE_TH_) ? s : 0.0f;
  flab[i] = (float)j;
}

// ==================== final NMS -> output (100 x 6) ====================
#define RCNN_N_ (RPN_KEEP_ * NUM_FG_)         // 900
__global__ void rcnn_nms_kernel(const float* __restrict__ fb, const float* __restrict__ fs,
                                const float* __restrict__ flab, float* __restrict__ out) {
  __shared__ float ls[RCNN_N_];
  __shared__ float rv[256];
  __shared__ int   ri[256];
  __shared__ float bbx[4];
  __shared__ int   curi;
  int tid = threadIdx.x;
  for (int j = tid; j < RCNN_N_; j += 256) ls[j] = fs[j];
  __syncthreads();
  for (int k = 0; k < FINAL_KEEP_; ++k) {
    float best = -1e30f; int bi = -1;
    for (int j = tid; j < RCNN_N_; j += 256) {
      float v = ls[j];
      if (v > best) { best = v; bi = j; }
    }
    rv[tid] = best; ri[tid] = bi;
    __syncthreads();
    for (int off = 128; off > 0; off >>= 1) {
      if (tid < off) {
        float vo = rv[tid + off]; int io = ri[tid + off];
        if (vo > rv[tid] || (vo == rv[tid] && io >= 0 && (ri[tid] < 0 || io < ri[tid]))) {
          rv[tid] = vo; ri[tid] = io;
        }
      }
      __syncthreads();
    }
    if (tid == 0) {
      int i = ri[0];
      bool ok = (i >= 0) && (rv[0] > 0.0f);
      curi = ok ? i : -1;
      if (ok) {
        bbx[0] = fb[i * 4 + 0]; bbx[1] = fb[i * 4 + 1];
        bbx[2] = fb[i * 4 + 2]; bbx[3] = fb[i * 4 + 3];
        out[k * 6 + 0] = bbx[0]; out[k * 6 + 1] = bbx[1];
        out[k * 6 + 2] = bbx[2]; out[k * 6 + 3] = bbx[3];
        out[k * 6 + 4] = ls[i];
        out[k * 6 + 5] = flab[i];
        ls[i] = -1.0f;
      } else {
        for (int c = 0; c < 6; ++c) out[k * 6 + c] = 0.0f;
      }
    }
    __syncthreads();
    int ci = curi;
    if (ci >= 0) {
      float bx1 = bbx[0], by1 = bbx[1], bx2 = bbx[2], by2 = bbx[3];
      float areai = fmaxf(bx2 - bx1 + 1.0f, 0.0f) * fmaxf(by2 - by1 + 1.0f, 0.0f);
      for (int j = tid; j < RCNN_N_; j += 256) {
        if (ls[j] <= 0.0f) continue;
        float jx1 = fb[j * 4 + 0], jy1 = fb[j * 4 + 1];
        float jx2 = fb[j * 4 + 2], jy2 = fb[j * 4 + 3];
        float areaj = fmaxf(jx2 - jx1 + 1.0f, 0.0f) * fmaxf(jy2 - jy1 + 1.0f, 0.0f);
        float ix1 = fmaxf(bx1, jx1), iy1 = fmaxf(by1, jy1);
        float ix2 = fminf(bx2, jx2), iy2 = fminf(by2, jy2);
        float inter = fmaxf(ix2 - ix1 + 1.0f, 0.0f) * fmaxf(iy2 - iy1 + 1.0f, 0.0f);
        float iou = inter / fmaxf(areai + areaj - inter, 1e-6f);
        if (iou > RCNN_NMS_TH_) ls[j] = -1.0f;
      }
    }
    __syncthreads();
  }
}

// ==================== host launcher ====================
static inline size_t align256(size_t x) { return (x + 255) & ~(size_t)255; }

extern "C" void kernel_launch(void* const* d_in, const int* in_sizes, int n_in,
                              void* d_out, int out_size, void* d_ws, size_t ws_size,
                              hipStream_t stream) {
  const float* rpn_cls  = (const float*)d_in[0];
  const float* rpn_bbox = (const float*)d_in[1];
  const float* ft       = (const float*)d_in[2];
  const float* w1       = (const float*)d_in[3];
  const float* b1       = (const float*)d_in[4];
  const float* w_cls    = (const float*)d_in[5];
  const float* b_cls    = (const float*)d_in[6];
  const float* w_bbox   = (const float*)d_in[7];
  const float* b_bbox   = (const float*)d_in[8];
  float* out = (float*)d_out;

  char* ws = (char*)d_ws;
  size_t cur = 0;
  float*    boxes   = (float*)(ws + cur);  cur = align256(cur + (size_t)NANCH * 4 * 4);
  float*    scores  = (float*)(ws + cur);  cur = align256(cur + (size_t)NANCH * 4);
  unsigned* hist    = (unsigned*)(ws + cur); cur = align256(cur + NBINS_ * 4);
  unsigned* cnt     = (unsigned*)(ws + cur); cur = align256(cur + 8 * 4);
  int*      meta    = (int*)(ws + cur);    cur = align256(cur + 8 * 4);
  float*    sel_b   = (float*)(ws + cur);  cur = align256(cur + PRE_NMS_ * 4 * 4);
  float*    sel_s   = (float*)(ws + cur);  cur = align256(cur + PRE_NMS_ * 4);
  float*    rois    = (float*)(ws + cur);  cur = align256(cur + RPN_KEEP_ * 4 * 4);
  float*    validf  = (float*)(ws + cur);  cur = align256(cur + RPN_KEEP_ * 4);
  float*    pooled  = (float*)(ws + cur);  cur = align256(cur + (size_t)RPN_KEEP_ * FEAT_DIM_ * 4);
  unsigned* Apack   = (unsigned*)(ws + cur); cur = align256(cur + (size_t)MT_ * KT_ * 32 * 8 * 4);
  unsigned* Bpack   = (unsigned*)(ws + cur); cur = align256(cur + (size_t)NT_ * KT_ * 32 * 8 * 4);
  float*    hbuf    = (float*)(ws + cur);  cur = align256(cur + (size_t)RPN_KEEP_ * HID_ * 4);
  float*    cls_sm  = (float*)(ws + cur);  cur = align256(cur + RPN_KEEP_ * 4 * 4);
  float*    bboxp   = (float*)(ws + cur);  cur = align256(cur + RPN_KEEP_ * 16 * 4);
  float*    fb      = (float*)(ws + cur);  cur = align256(cur + RCNN_N_ * 4 * 4);
  float*    fsb     = (float*)(ws + cur);  cur = align256(cur + RCNN_N_ * 4);
  float*    flab    = (float*)(ws + cur);  cur = align256(cur + RCNN_N_ * 4);
  (void)cur; (void)ws_size; (void)in_sizes; (void)n_in; (void)out_size;

  // per-call zeroing of selection state (graph-capturable memset nodes)
  hipMemsetAsync(hist, 0, NBINS_ * 4, stream);
  hipMemsetAsync(cnt, 0, 8 * 4, stream);
  hipMemsetAsync(sel_b, 0, PRE_NMS_ * 4 * 4, stream);
  hipMemsetAsync(sel_s, 0, PRE_NMS_ * 4, stream);

  rpn_decode_kernel<<<(NANCH + 255) / 256, 256, 0, stream>>>(rpn_cls, rpn_bbox, boxes, scores);
  hist_kernel<<<(NANCH + 255) / 256, 256, 0, stream>>>(scores, hist);
  thresh_kernel<<<1, 32, 0, stream>>>(hist, meta);
  compact_kernel<<<(NANCH + 255) / 256, 256, 0, stream>>>(scores, boxes, meta, cnt, sel_s, sel_b);
  rpn_nms_kernel<<<1, 256, 0, stream>>>(sel_b, sel_s, rois, validf);
  psroi_kernel<<<(RPN_KEEP_ * FEAT_DIM_ + 255) / 256, 256, 0, stream>>>(ft, rois, validf, pooled);
  packA_kernel<<<(MT_ * KT_ * 32 * 8) / 256, 256, 0, stream>>>(pooled, Apack);
  packB_kernel<<<(NT_ * KT_ * 32 * 8) / 256, 256, 0, stream>>>(w1, Bpack);
  gemm_fc1_kernel<<<(MT_ * NT_) / 8, 256, 0, stream>>>(Apack, Bpack, b1, hbuf);
  fc2_kernel<<<RPN_KEEP_, 256, 0, stream>>>(hbuf, w_cls, b_cls, w_bbox, b_bbox, cls_sm, bboxp);
  rcnn_decode_kernel<<<(RCNN_N_ + 255) / 256, 256, 0, stream>>>(rois, validf, cls_sm, bboxp, fb, fsb, flab);
  rcnn_nms_kernel<<<1, 256, 0, stream>>>(fb, fsb, flab, out);
}
